// MuSc_36584531427417
// MI455X (gfx1250) — matrix operand order repaired and checked by reading, not compiled, not run on hardware
//
#include <hip/hip_runtime.h>
#include <hip/hip_bf16.h>

// ---------------- problem constants ----------------
constexpr int LL = 4, BB = 8, PP = 1369, DD = 1024;
constexpr int PH = 37, PW = 37, HH = 518, WW = 518;
constexpr int NB = BB - 1;          // 7 neighbors
constexpr int RR = 3;               // radii {1,3,5}
constexpr int ITILES = 11;          // ceil(1369/128)
constexpr int CSTRIDE = 40;         // ushorts per staged column (64B data + 16B pad)

// ---------------- output layout (flat f32, concat in return order) --------
constexpr size_t OFF_FS   = 0;                                   // (8,)
constexpr size_t OFF_PIX  = 8;                                   // (8,518,518)
constexpr size_t OFF_MIN  = OFF_PIX + (size_t)BB * HH * WW;      // (8,4,3,7,1369)
constexpr size_t OFF_MAXI = OFF_MIN + (size_t)BB * LL * RR * NB * PP; // (8,)
constexpr size_t OFF_TKI  = OFF_MAXI + BB;                       // (8,4,3,3,1369)
constexpr size_t OFF_TKS  = OFF_TKI + (size_t)BB * LL * RR * 3 * PP;

// ---------------- workspace layout (bytes) ----------------
constexpr size_t NTOT   = (size_t)LL * BB * PP * DD;             // 44,859,392 elems
constexpr size_t WS_FN  = 0;                                     // bf16 fn
constexpr size_t WS_RF  = WS_FN + NTOT * 2;                      // bf16 rf
constexpr size_t WS_X2  = WS_RF + NTOT * 2;                      // f32 (L,B,P)
constexpr size_t WS_MD  = WS_X2 + (size_t)LL * BB * PP * 4;      // f32 (L,B,P,7)
constexpr size_t WS_MI  = WS_MD + (size_t)LL * BB * PP * NB * 4; // i32 (L,B,P,7)
constexpr size_t WS_SC  = WS_MI + (size_t)LL * BB * PP * NB * 4; // f32 (B,P)

// ---------------- vector types for WMMA ----------------
typedef __attribute__((ext_vector_type(16))) __bf16 v16bf;
typedef __attribute__((ext_vector_type(8)))  __bf16 v8bf;
typedef __attribute__((ext_vector_type(8)))  float  v8f;

__device__ __forceinline__ unsigned short f2bf(float f) {
    union { float f; unsigned int u; } v; v.f = f;
    unsigned int u = v.u;
    unsigned int r = u + 0x7fffu + ((u >> 16) & 1u);   // round-to-nearest-even
    return (unsigned short)(r >> 16);
}
__device__ __forceinline__ float bf2f(unsigned short u) {
    union { unsigned int u; float f; } v; v.u = ((unsigned int)u) << 16;
    return v.f;
}

// ---------------- zero scores ----------------
__global__ void k_zero(float* __restrict__ sc) {
    int i = blockIdx.x * blockDim.x + threadIdx.x;
    if (i < BB * PP) sc[i] = 0.f;
}

// ---------------- layernorm over D, store bf16 ----------------
__global__ __launch_bounds__(256) void k_ln(const float* __restrict__ feat,
                                            unsigned short* __restrict__ fn) {
    __shared__ float red[256];
    const int row = blockIdx.x;                 // (l*B+b)*P + p
    const int t = threadIdx.x;
    const float* x = feat + (size_t)row * DD;
    float s = 0.f, sq = 0.f;
    for (int d = t; d < DD; d += 256) { float v = x[d]; s += v; sq += v * v; }
    red[t] = s; __syncthreads();
    for (int o = 128; o > 0; o >>= 1) { if (t < o) red[t] += red[t + o]; __syncthreads(); }
    const float mean = red[0] / DD; __syncthreads();
    red[t] = sq; __syncthreads();
    for (int o = 128; o > 0; o >>= 1) { if (t < o) red[t] += red[t + o]; __syncthreads(); }
    const float var = red[0] / DD - mean * mean;
    const float rstd = rsqrtf(fmaxf(var, 0.f) + 1e-5f);
    for (int d = t; d < DD; d += 256)
        fn[(size_t)row * DD + d] = f2bf((x[d] - mean) * rstd);
}

// ---------------- r x r box average (SAME, zero pad), + squared norms ------
__global__ __launch_bounds__(256) void k_box(const unsigned short* __restrict__ fn,
                                             unsigned short* __restrict__ rf,
                                             float* __restrict__ x2, int r) {
    __shared__ float red[256];
    const int row = blockIdx.x;                 // (l*B+b)*P + p
    const int p = row % PP, lb = row / PP;
    const int ph = p / PW, pw = p % PW;
    const int t = threadIdx.x;
    const int h = (r - 1) / 2;
    const float inv = 1.0f / (float)(r * r);
    float sq = 0.f;
    for (int d = t; d < DD; d += 256) {
        float acc = 0.f;
        for (int dy = -h; dy <= h; ++dy) {
            int py = ph + dy; if (py < 0 || py >= PH) continue;
            for (int dx = -h; dx <= h; ++dx) {
                int px = pw + dx; if (px < 0 || px >= PW) continue;
                acc += bf2f(fn[((size_t)lb * PP + py * PW + px) * DD + d]);
            }
        }
        unsigned short q = f2bf(acc * inv);
        rf[(size_t)row * DD + d] = q;
        float qv = bf2f(q);
        sq += qv * qv;
    }
    red[t] = sq; __syncthreads();
    for (int o = 128; o > 0; o >>= 1) { if (t < o) red[t] += red[t + o]; __syncthreads(); }
    if (t == 0) x2[row] = red[0];
}

// ---------------- WMMA distance + min/argmin over q ----------------
// grid: (11 i-tiles, b, l*B + c); block 256 = 8 waves; wave w owns rows
// [it*128 + w*16, +16) vs all 1369 columns of image c. The 128x32 B sub-panel
// for each K-step is staged once per block into double-buffered LDS (80B column
// stride -> conflict-free fragment reads), eliminating the 8x cross-wave
// redundancy in L2 traffic that otherwise dominates over the WMMA time.
__global__ __launch_bounds__(256) void k_msm(const unsigned short* __restrict__ rf,
                                             const float* __restrict__ x2,
                                             float* __restrict__ md_ns,
                                             int* __restrict__ mi_ns) {
    const int it = blockIdx.x;
    const int b  = blockIdx.y;
    const int l  = blockIdx.z / BB;
    const int c  = blockIdx.z % BB;
    if (c == b) return;

    __shared__ unsigned short ldsB[2][128 * CSTRIDE];   // 2 x 10 KB

    const int t    = threadIdx.x;
    const int wv   = t >> 5;
    const int lane = t & 31;
    const int nl   = lane & 15;
    const int hi   = lane >> 4;
    const int scol  = t >> 1;       // staging: column 0..127
    const int shalf = t & 1;        // staging: which 32B half of the 64B column chunk

    const unsigned short* Fb = rf + (size_t)((l * BB + b) * PP) * DD;
    const unsigned short* Fc = rf + (size_t)((l * BB + c) * PP) * DD;
    const float* x2b = x2 + (l * BB + b) * PP;
    const float* x2c = x2 + (l * BB + c) * PP;

    const int p_base = it * 128 + wv * 16;

    // A-matrix row this lane supplies (16x32 bf16 layout: lanes 0-15 & 16-31 both M=0..15)
    int arow = p_base + nl; if (arow >= PP) arow = PP - 1;
    const unsigned short* Arow = Fb + (size_t)arow * DD;

    // squared norms of the 8 output rows this lane accumulates (m = e + hi*8)
    float x2r[8];
#pragma unroll
    for (int e = 0; e < 8; ++e) {
        int p = p_base + e + hi * 8; if (p >= PP) p = PP - 1;
        x2r[e] = x2b[p];
    }

    float minv[8]; int mini[8];
#pragma unroll
    for (int e = 0; e < 8; ++e) { minv[e] = __builtin_inff(); mini[e] = 0; }

    for (int qt = 0; qt < ITILES; ++qt) {
        const int q_base = qt * 128;

        // column this thread stages (clamped; junk columns are masked at fold time)
        int jc = q_base + scol; if (jc >= PP) jc = PP - 1;
        const unsigned short* srcCol = Fc + (size_t)jc * DD + shalf * 16;

        v8f acc[8];
#pragma unroll
        for (int s = 0; s < 8; ++s) acc[s] = {};

        // prologue: stage K-step 0 into buffer 0
        {
            const uint4* src = (const uint4*)srcCol;
            uint4* dst = (uint4*)&ldsB[0][scol * CSTRIDE + shalf * 16];
            dst[0] = src[0]; dst[1] = src[1];
        }

        for (int ki = 0; ki < DD / 32; ++ki) {
            __syncthreads();   // staged buffer (ki&1) ready; prior reads of other buf done
            if (ki + 1 < DD / 32) {
                const uint4* src = (const uint4*)(srcCol + (ki + 1) * 32);
                uint4* dst = (uint4*)&ldsB[(ki + 1) & 1][scol * CSTRIDE + shalf * 16];
                dst[0] = src[0]; dst[1] = src[1];
            }
            const int kb = ki * 32;
            // A fragment: elem i -> K = kb + hi*16 + i
            v16bf af = *(const v16bf*)(Arow + kb + hi * 16);

            const unsigned short* buf = ldsB[ki & 1];
#pragma unroll
            for (int s = 0; s < 8; ++s) {
                // B fragment: col = s*16+nl, elem i -> K = kb + hi*16 + i
                union { uint4 q[2]; v16bf v; } u;
                const uint4* bp = (const uint4*)&buf[(s * 16 + nl) * CSTRIDE + hi * 16];
                u.q[0] = bp[0]; u.q[1] = bp[1];
                acc[s] = __builtin_amdgcn_wmma_f32_16x16x32_bf16(
                    false, af, false, u.v, (short)0, acc[s], false, false);
            }
        }
        __syncthreads();       // all reads of last buffer done before next qt restages

        // fold distances into running per-row (min, argmin) over columns j
#pragma unroll
        for (int s = 0; s < 8; ++s) {
            const int j = q_base + s * 16 + nl;
            if (j < PP) {
                const float x2j = x2c[j];
#pragma unroll
                for (int e = 0; e < 8; ++e) {
                    float d2 = x2r[e] + x2j - 2.0f * acc[s][e];
                    float d = sqrtf(fmaxf(d2, 1e-12f));
                    if (d < minv[e]) { minv[e] = d; mini[e] = j; }
                }
            }
        }
    }

    // cross-lane reduce over the 16 columns held in each half-wave
#pragma unroll
    for (int off = 1; off < 16; off <<= 1) {
#pragma unroll
        for (int e = 0; e < 8; ++e) {
            float ov = __shfl_xor(minv[e], off, 32);
            int   oi = __shfl_xor(mini[e], off, 32);
            if (ov < minv[e] || (ov == minv[e] && oi < mini[e])) { minv[e] = ov; mini[e] = oi; }
        }
    }

    if (nl == 0) {                       // lane 0 (rows 0..7) and lane 16 (rows 8..15)
        const int nidx = c - (c > b ? 1 : 0);
#pragma unroll
        for (int e = 0; e < 8; ++e) {
            int p = p_base + e + hi * 8;
            if (p < PP) {
                size_t o = ((size_t)((l * BB + b) * PP + p)) * NB + nidx;
                md_ns[o] = minv[e];
                mi_ns[o] = mini[e];
            }
        }
    }
}

// ---------------- stable 3-smallest selection + output writes ----------------
__global__ void k_post(const float* __restrict__ md_ns, const int* __restrict__ mi_ns,
                       float* __restrict__ scores, float* __restrict__ out, int r_i) {
    const int idx = blockIdx.x * blockDim.x + threadIdx.x;
    if (idx >= BB * PP) return;
    const int bimg = idx / PP, p = idx % PP;
    float* out_min = out + OFF_MIN;
    float* out_tki = out + OFF_TKI;
    float* out_tks = out + OFF_TKS;
    float accum = 0.f;
    for (int l = 0; l < LL; ++l) {
        const size_t base = ((size_t)((l * BB + bimg) * PP + p)) * NB;
        float v[NB]; int mi[NB];
        for (int n = 0; n < NB; ++n) { v[n] = md_ns[base + n]; mi[n] = mi_ns[base + n]; }
        for (int n = 0; n < NB; ++n)
            out_min[((((size_t)bimg * LL + l) * RR + r_i) * NB + n) * PP + p] = (float)mi[n];
        bool used[NB] = {};
        float s3 = 0.f;
        for (int j = 0; j < 3; ++j) {
            int best = -1; float bv = __builtin_inff();
            for (int n = 0; n < NB; ++n)
                if (!used[n] && v[n] < bv) { bv = v[n]; best = n; }
            if (best < 0) best = j;      // degenerate (NaN) fallback
            used[best] = true;
            out_tki[((((size_t)bimg * LL + l) * RR + r_i) * 3 + j) * PP + p] = (float)best;
            out_tks[((((size_t)bimg * LL + l) * RR + r_i) * 3 + j) * PP + p] = v[best];
            s3 += v[best];
        }
        accum += s3 * (1.0f / 3.0f);
    }
    scores[idx] += accum * (1.0f / (LL * RR));   // mean over l then over r
}

// ---------------- per-image max / argmax (first occurrence) ----------------
__global__ __launch_bounds__(256) void k_final(const float* __restrict__ scores,
                                               float* __restrict__ out) {
    __shared__ float rv[256]; __shared__ int ri[256];
    const int bimg = blockIdx.x, t = threadIdx.x;
    float mv = -__builtin_inff(); int mi = PP;
    for (int p = t; p < PP; p += 256) {
        float v = scores[bimg * PP + p];
        if (v > mv || (v == mv && p < mi)) { mv = v; mi = p; }
    }
    rv[t] = mv; ri[t] = mi; __syncthreads();
    for (int o = 128; o > 0; o >>= 1) {
        if (t < o) {
            if (rv[t + o] > rv[t] || (rv[t + o] == rv[t] && ri[t + o] < ri[t])) {
                rv[t] = rv[t + o]; ri[t] = ri[t + o];
            }
        }
        __syncthreads();
    }
    if (t == 0) { out[OFF_FS + bimg] = rv[0]; out[OFF_MAXI + bimg] = (float)ri[0]; }
}

// ---------------- bilinear upsample 37x37 -> 518x518 ----------------
__global__ void k_up(const float* __restrict__ scores, float* __restrict__ out) {
    const int idx = blockIdx.x * blockDim.x + threadIdx.x;
    if (idx >= BB * HH * WW) return;
    const int bimg = idx / (HH * WW);
    const int rem = idx % (HH * WW);
    const int hh = rem / WW, ww = rem % WW;
    const float posh = (float)hh * (float)(PH - 1) / (float)(HH - 1);
    int i0 = (int)posh; float fh = posh - (float)i0;
    int i1 = i0 + 1 < PH ? i0 + 1 : PH - 1;
    const float posw = (float)ww * (float)(PW - 1) / (float)(WW - 1);
    int j0 = (int)posw; float fw = posw - (float)j0;
    int j1 = j0 + 1 < PW ? j0 + 1 : PW - 1;
    const float* pb = scores + bimg * PP;
    float p00 = pb[i0 * PW + j0], p01 = pb[i0 * PW + j1];
    float p10 = pb[i1 * PW + j0], p11 = pb[i1 * PW + j1];
    float top = p00 + fw * (p01 - p00);
    float bot = p10 + fw * (p11 - p10);
    out[OFF_PIX + (size_t)idx] = top + fh * (bot - top);
}

// ---------------- host orchestration ----------------
extern "C" void kernel_launch(void* const* d_in, const int* in_sizes, int n_in,
                              void* d_out, int out_size, void* d_ws, size_t ws_size,
                              hipStream_t stream) {
    (void)in_sizes; (void)n_in; (void)out_size; (void)ws_size;
    const float* feat = (const float*)d_in[0];
    float* out = (float*)d_out;
    char* ws = (char*)d_ws;

    unsigned short* fn = (unsigned short*)(ws + WS_FN);
    unsigned short* rf = (unsigned short*)(ws + WS_RF);
    float* x2 = (float*)(ws + WS_X2);
    float* md = (float*)(ws + WS_MD);
    int*   mi = (int*)(ws + WS_MI);
    float* sc = (float*)(ws + WS_SC);

    k_zero<<<(BB * PP + 255) / 256, 256, 0, stream>>>(sc);
    k_ln<<<LL * BB * PP, 256, 0, stream>>>(feat, fn);

    const int RLIST[RR] = {1, 3, 5};
    for (int ri = 0; ri < RR; ++ri) {
        k_box<<<LL * BB * PP, 256, 0, stream>>>(fn, rf, x2, RLIST[ri]);
        dim3 grid(ITILES, BB, LL * BB);
        k_msm<<<grid, 256, 0, stream>>>(rf, x2, md, mi);
        k_post<<<(BB * PP + 255) / 256, 256, 0, stream>>>(md, mi, sc, out, ri);
    }
    k_final<<<BB, 256, 0, stream>>>(sc, out);
    k_up<<<(BB * HH * WW + 255) / 256, 256, 0, stream>>>(sc, out);
}
